// LSTMDecoderLayer_60790967107736
// MI455X (gfx1250) — matrix operand order repaired and checked by reading, not compiled, not run on hardware
//
#include <hip/hip_runtime.h>
#include <stdint.h>

// ---------------------------------------------------------------------------
// LSTM decoder layer for MI455X (gfx1250, wave32, WMMA bf16).
//   T=512, B=64, E=H=1024.
// Strategy:
//  * bf16 weights (42MB, resident in 192MB L2), fp32 accumulate via
//    v_wmma_f32_16x16x32_bf16
//  * per timestep: 2 fused GEMM+LSTM-cell kernels (32 WGs each, 8 waves/WG),
//    software-pipelined A-panel staging (register prefetch over the barrier)
//  * final: batched projection GEMM (+bias+residual) then LayerNorm
// ---------------------------------------------------------------------------

typedef __bf16 v8bf  __attribute__((ext_vector_type(8)));
typedef __bf16 v16bf __attribute__((ext_vector_type(16)));
typedef float  v8f   __attribute__((ext_vector_type(8)));

#define TT 512
#define BB 64
#define HH 1024
#define EE 1024
#define AP 72   // LDS A-panel row stride in bf16 (64 + 8 pad -> 144B rows)

// ---------------- small utility kernels ----------------

__global__ void cvt_f32_to_bf16(const float* __restrict__ s, __bf16* __restrict__ d, int n) {
    int i = blockIdx.x * blockDim.x + threadIdx.x;
    int stride = gridDim.x * blockDim.x;
    for (; i < n; i += stride) d[i] = (__bf16)s[i];
}

__global__ void add_f32(const float* __restrict__ a, const float* __restrict__ b,
                        float* __restrict__ d, int n) {
    int i = blockIdx.x * blockDim.x + threadIdx.x;
    if (i < n) d[i] = a[i] + b[i];
}

__global__ void zero_u32(uint32_t* __restrict__ p, int n) {
    int i = blockIdx.x * blockDim.x + threadIdx.x;
    int stride = gridDim.x * blockDim.x;
    for (; i < n; i += stride) p[i] = 0u;
}

// ---------------- fragment helpers (contiguous k order, same for A and B) ----------

// A 16x32 bf16 tile from LDS: lane L -> row L&15; 16 contiguous K at (L>>4)*16.
__device__ inline v16bf load_a_frag(const __bf16* base, int lane, int kofs) {
    int r  = lane & 15;
    int ko = (lane >> 4) << 4;
    const __bf16* p = base + r * AP + kofs + ko;
    union { v16bf v; v8bf h[2]; } u;
    u.h[0] = *(const v8bf*)(p);
    u.h[1] = *(const v8bf*)(p + 8);
    return u.v;
}

__device__ inline float fast_sigmoid(float x) { return 1.0f / (1.0f + __expf(-x)); }
__device__ inline float fast_tanh(float x)    { return 2.0f / (1.0f + __expf(-2.0f * x)) - 1.0f; }

// Staging register block: one thread carries 16 columns of its row of the
// next A chunk across the barrier (either 16 fp32 to convert, or 16 bf16).
union StageReg {
    float4 f4[4];   // fp32 x_t region
    v16bf  bv;      // bf16 activation regions
};

__device__ inline void prefetch_stage(StageReg& r, const float* xf, int xK,
                                      const __bf16* aB0, const __bf16* aB1,
                                      int srow, int kk) {
    if (kk < xK) {
        const float4* s4 = (const float4*)(xf + (size_t)srow * 1024 + kk);
#pragma unroll
        for (int i = 0; i < 4; ++i) r.f4[i] = s4[i];
    } else {
        int kb = kk - xK;
        const __bf16* src = (kb < 1024 ? aB0 + (size_t)srow * 1024 + kb
                                       : aB1 + (size_t)srow * 1024 + (kb - 1024));
        r.bv = *(const v16bf*)src;
    }
}

__device__ inline void store_stage(const StageReg& r, __bf16* dst, bool isF32) {
    if (isF32) {
#pragma unroll
        for (int i = 0; i < 4; ++i) {
            float4 v = r.f4[i];
            dst[4 * i + 0] = (__bf16)v.x;
            dst[4 * i + 1] = (__bf16)v.y;
            dst[4 * i + 2] = (__bf16)v.z;
            dst[4 * i + 3] = (__bf16)v.w;
        }
    } else {
        *(v16bf*)dst = r.bv;
    }
}

// ---------------- fused GEMM + LSTM cell ----------------
// gates[64 x 4H] = A[64 x KTOT] * Wcat^T + bsum, then cell update for this
// block's 32-wide j-slice across all four gates.
// A regions along K: [0,xK): fp32 x_t ; then two 1024-wide bf16 regions.
// W regions along K: [0,wsplit): W0 (row stride ws0) ; rest: W1 (stride ws1).
__global__ __launch_bounds__(256)
void lstm_step_kernel(const float* __restrict__ xf,      // fp32 A region (or null)
                      int xK,                            // size of fp32 region (0 or 1024)
                      const __bf16* __restrict__ aB0,    // bf16 A region 0 (1024 wide)
                      const __bf16* __restrict__ aB1,    // bf16 A region 1 (1024 wide)
                      int KTOT,
                      const __bf16* __restrict__ W0, int ws0, int wsplit,
                      const __bf16* __restrict__ W1, int ws1,
                      const float* __restrict__ bsum,    // [4H] combined bias
                      float* __restrict__ c_state,       // [B*H] fp32, in-place
                      __bf16* __restrict__ hA,           // [B*H] bf16 recurrent out
                      __bf16* __restrict__ hB)           // optional 2nd copy (outs[t])
{
    __shared__ __bf16 As[64 * AP];          // 9.0 KB staged A panel (64 x 64 bf16)
    __shared__ float  Gs[4 * 64 * 32];      // 32 KB gate exchange buffer

    const int tid  = threadIdx.x;
    const int wave = tid >> 5;
    const int lane = tid & 31;
    const int g    = wave >> 1;                                  // gate 0..3
    const int n0   = g * HH + blockIdx.x * 32 + (wave & 1) * 16; // output col base
    const int nW   = n0 + (lane & 15);                           // this lane's W row

    v8f acc[4];
#pragma unroll
    for (int mt = 0; mt < 4; ++mt) acc[mt] = (v8f)0.0f;

    const int srow = tid >> 2;          // 0..63 staging row
    const int sq   = tid & 3;           // 16-col quad

    // B-fragment loader: lane L -> W row n, 32 contiguous bytes at this lane's
    // K sub-chunk (same contiguous k order as the A fragment).
    auto load_b = [&](int kc, int ks) -> v16bf {
        int kf = kc + ks * 32 + ((lane >> 4) << 4);
        const __bf16* wp = (kf < wsplit)
            ? (W0 + (size_t)nW * ws0 + kf)
            : (W1 + (size_t)nW * ws1 + (kf - wsplit));
        return *(const v16bf*)wp;
    };

    // ---- software pipeline: prefetch chunk 0 (A panel + both B frags)
    StageReg sreg;
    prefetch_stage(sreg, xf, xK, aB0, aB1, srow, sq * 16);
    v16bf bcur0 = load_b(0, 0);
    v16bf bcur1 = load_b(0, 1);

    for (int k0 = 0; k0 < KTOT; k0 += 64) {
        __syncthreads();                              // prior chunk's reads done
        store_stage(sreg, As + srow * AP + sq * 16, (k0 < xK));
        __syncthreads();                              // panel visible

        // prefetch next chunk while computing this one (hides L2 latency)
        v16bf bnxt0 = bcur0, bnxt1 = bcur1;
        int kn = k0 + 64;
        if (kn < KTOT) {
            prefetch_stage(sreg, xf, xK, aB0, aB1, srow, kn + sq * 16);
            bnxt0 = load_b(kn, 0);
            bnxt1 = load_b(kn, 1);
        }

        // ---- 2 WMMA k-steps over the staged chunk
#pragma unroll
        for (int ks = 0; ks < 2; ++ks) {
            v16bf bfrag = ks ? bcur1 : bcur0;
#pragma unroll
            for (int mt = 0; mt < 4; ++mt) {
                v16bf afrag = load_a_frag(As + mt * 16 * AP, lane, ks * 32);
                acc[mt] = __builtin_amdgcn_wmma_f32_16x16x32_bf16(
                    false, afrag, false, bfrag, (short)0, acc[mt], false, false);
            }
        }
        bcur0 = bnxt0;
        bcur1 = bnxt1;
    }

    // ---- scatter gate accumulators to LDS: Gs[gate][b][j'] (j' in 0..31)
    {
        int jcol = (wave & 1) * 16 + (lane & 15);
#pragma unroll
        for (int mt = 0; mt < 4; ++mt) {
#pragma unroll
            for (int v = 0; v < 8; ++v) {
                int b = mt * 16 + v + ((lane >> 4) << 3);
                Gs[g * 2048 + b * 32 + jcol] = acc[mt][v];
            }
        }
    }
    __syncthreads();

    // ---- LSTM cell elementwise: 64 x 32 outputs, 8 per thread
    {
        int b  = tid >> 2;
        int jb = (tid & 3) * 8;
#pragma unroll
        for (int jj = 0; jj < 8; ++jj) {
            int jc = jb + jj;
            int n  = blockIdx.x * 32 + jc;            // column within H
            float gi = Gs[0 * 2048 + b * 32 + jc] + bsum[0 * HH + n];
            float gf = Gs[1 * 2048 + b * 32 + jc] + bsum[1 * HH + n];
            float gg = Gs[2 * 2048 + b * 32 + jc] + bsum[2 * HH + n];
            float go = Gs[3 * 2048 + b * 32 + jc] + bsum[3 * HH + n];
            float iv = fast_sigmoid(gi);
            float fv = fast_sigmoid(gf);
            float gv = fast_tanh(gg);
            float ov = fast_sigmoid(go);
            size_t idx = (size_t)b * HH + n;
            float c = fv * c_state[idx] + iv * gv;
            c_state[idx] = c;
            float h = ov * fast_tanh(c);
            __bf16 hb = (__bf16)h;
            hA[idx] = hb;
            if (hB) hB[idx] = hb;
        }
    }
}

// ---------------- projection GEMM: y = outs @ Wlin^T + b + residual ----------
__global__ __launch_bounds__(256)
void proj_kernel(const __bf16* __restrict__ outsb,   // [T*B x 1024] bf16
                 const __bf16* __restrict__ Wlin,    // [1024 x 1024] bf16 [N][K]
                 const float*  __restrict__ blin,    // [1024]
                 const float*  __restrict__ x,       // [T*B x 1024] residual
                 float* __restrict__ y)               // [T*B x 1024] out
{
    __shared__ __bf16 As[64 * AP];

    const int t    = blockIdx.y;                     // timestep = 64-row M block
    const int tid  = threadIdx.x;
    const int wave = tid >> 5;
    const int lane = tid & 31;
    const int n0   = blockIdx.x * 128 + wave * 16;
    const int nW   = n0 + (lane & 15);
    const __bf16* A = outsb + (size_t)t * BB * HH;

    v8f acc[4];
#pragma unroll
    for (int mt = 0; mt < 4; ++mt) acc[mt] = (v8f)0.0f;

    const int srow = tid >> 2, sq = tid & 3;

    auto load_b = [&](int kc, int ks) -> v16bf {
        int kf = kc + ks * 32 + ((lane >> 4) << 4);
        return *(const v16bf*)(Wlin + (size_t)nW * HH + kf);
    };

    v16bf sreg  = *(const v16bf*)(A + (size_t)srow * HH + sq * 16);
    v16bf bcur0 = load_b(0, 0);
    v16bf bcur1 = load_b(0, 1);

    for (int k0 = 0; k0 < HH; k0 += 64) {
        __syncthreads();
        *(v16bf*)(As + srow * AP + sq * 16) = sreg;
        __syncthreads();

        v16bf bnxt0 = bcur0, bnxt1 = bcur1;
        int kn = k0 + 64;
        if (kn < HH) {
            sreg  = *(const v16bf*)(A + (size_t)srow * HH + kn + sq * 16);
            bnxt0 = load_b(kn, 0);
            bnxt1 = load_b(kn, 1);
        }

#pragma unroll
        for (int ks = 0; ks < 2; ++ks) {
            v16bf bfrag = ks ? bcur1 : bcur0;
#pragma unroll
            for (int mt = 0; mt < 4; ++mt) {
                v16bf afrag = load_a_frag(As + mt * 16 * AP, lane, ks * 32);
                acc[mt] = __builtin_amdgcn_wmma_f32_16x16x32_bf16(
                    false, afrag, false, bfrag, (short)0, acc[mt], false, false);
            }
        }
        bcur0 = bnxt0;
        bcur1 = bnxt1;
    }

    int n = n0 + (lane & 15);
    float bl = blin[n];
#pragma unroll
    for (int mt = 0; mt < 4; ++mt) {
#pragma unroll
        for (int v = 0; v < 8; ++v) {
            int b = mt * 16 + v + ((lane >> 4) << 3);
            size_t idx = ((size_t)t * BB + b) * HH + n;
            y[idx] = acc[mt][v] + bl + x[idx];
        }
    }
}

// ---------------- LayerNorm (one wave per 1024-wide row) ----------------
__global__ __launch_bounds__(256)
void ln_kernel(float* __restrict__ y, const float* __restrict__ gamma,
               const float* __restrict__ beta, int rows)
{
    int row  = blockIdx.x * 8 + (threadIdx.x >> 5);
    int lane = threadIdx.x & 31;
    if (row >= rows) return;
    float* p = y + (size_t)row * EE;

    float s = 0.0f, s2 = 0.0f;
    for (int k = lane; k < EE; k += 32) { float v = p[k]; s += v; s2 += v * v; }
#pragma unroll
    for (int off = 16; off > 0; off >>= 1) {
        s  += __shfl_xor(s,  off, 32);
        s2 += __shfl_xor(s2, off, 32);
    }
    float mean = s * (1.0f / EE);
    float var  = s2 * (1.0f / EE) - mean * mean;
    float inv  = rsqrtf(var + 1e-5f);
    for (int k = lane; k < EE; k += 32)
        p[k] = (p[k] - mean) * inv * gamma[k] + beta[k];
}

// ---------------- host-side orchestration ----------------

extern "C" void kernel_launch(void* const* d_in, const int* in_sizes, int n_in,
                              void* d_out, int out_size, void* d_ws, size_t ws_size,
                              hipStream_t stream) {
    const float* x      = (const float*)d_in[0];
    const float* W_ih0  = (const float*)d_in[1];
    const float* b_ih0  = (const float*)d_in[2];
    const float* W_hh0  = (const float*)d_in[3];
    const float* b_hh0  = (const float*)d_in[4];
    const float* W_ih1  = (const float*)d_in[5];
    const float* b_ih1  = (const float*)d_in[6];
    const float* W_hh1  = (const float*)d_in[7];
    const float* b_hh1  = (const float*)d_in[8];
    const float* W_lin  = (const float*)d_in[9];
    const float* b_lin  = (const float*)d_in[10];
    const float* ln_g   = (const float*)d_in[11];
    const float* ln_b   = (const float*)d_in[12];
    float* y = (float*)d_out;

    // ---- workspace carve-up (all offsets 256B aligned) ----
    size_t off = 0;
    auto carve = [&](size_t bytes) -> void* {
        void* p = (char*)d_ws + off;
        off += (bytes + 255) & ~(size_t)255;
        return p;
    };
    __bf16* Wih0b = (__bf16*)carve((size_t)4 * HH * (EE + HH) * 2); // [4096 x 2048]
    __bf16* Whh0b = (__bf16*)carve((size_t)4 * HH * HH * 2);        // [4096 x 1024]
    __bf16* Wih1b = (__bf16*)carve((size_t)4 * HH * HH * 2);
    __bf16* Whh1b = (__bf16*)carve((size_t)4 * HH * HH * 2);
    __bf16* Wlinb = (__bf16*)carve((size_t)EE * HH * 2);            // [1024 x 1024]
    float*  bsum0 = (float*)carve((size_t)4 * HH * 4);
    float*  bsum1 = (float*)carve((size_t)4 * HH * 4);
    __bf16* h0b   = (__bf16*)carve((size_t)BB * HH * 2);
    __bf16* h1b   = (__bf16*)carve((size_t)BB * HH * 2);
    float*  c0    = (float*)carve((size_t)BB * HH * 4);
    float*  c1    = (float*)carve((size_t)BB * HH * 4);
    __bf16* outsb = (__bf16*)carve((size_t)TT * BB * HH * 2);       // 67 MB
    (void)ws_size; (void)in_sizes; (void)n_in; (void)out_size;

    // ---- 1) weight conversion to bf16 + bias pre-sum (every call; deterministic)
    cvt_f32_to_bf16<<<2048, 256, 0, stream>>>(W_ih0, Wih0b, 4 * HH * (EE + HH));
    cvt_f32_to_bf16<<<2048, 256, 0, stream>>>(W_hh0, Whh0b, 4 * HH * HH);
    cvt_f32_to_bf16<<<2048, 256, 0, stream>>>(W_ih1, Wih1b, 4 * HH * HH);
    cvt_f32_to_bf16<<<2048, 256, 0, stream>>>(W_hh1, Whh1b, 4 * HH * HH);
    cvt_f32_to_bf16<<<1024, 256, 0, stream>>>(W_lin, Wlinb, EE * HH);
    add_f32<<<(4 * HH + 255) / 256, 256, 0, stream>>>(b_ih0, b_hh0, bsum0, 4 * HH);
    add_f32<<<(4 * HH + 255) / 256, 256, 0, stream>>>(b_ih1, b_hh1, bsum1, 4 * HH);

    // ---- 2) zero recurrent state (h0, h1/feed, c0, c1)
    zero_u32<<<64, 256, 0, stream>>>((uint32_t*)h0b, BB * HH / 2);
    zero_u32<<<64, 256, 0, stream>>>((uint32_t*)h1b, BB * HH / 2);
    zero_u32<<<64, 256, 0, stream>>>((uint32_t*)c0,  BB * HH);
    zero_u32<<<64, 256, 0, stream>>>((uint32_t*)c1,  BB * HH);

    // ---- 3) recurrence: two fused GEMM+cell kernels per timestep
    for (int t = 0; t < TT; ++t) {
        const float* x_t = x + (size_t)t * BB * EE;
        // layer 0: A = [x_t (fp32) | feed=h1 | h0], W = [W_ih0 | W_hh0]
        lstm_step_kernel<<<32, 256, 0, stream>>>(
            x_t, /*xK=*/EE, /*aB0=*/h1b, /*aB1=*/h0b, /*KTOT=*/EE + 2 * HH,
            Wih0b, /*ws0=*/EE + HH, /*wsplit=*/EE + HH, Whh0b, /*ws1=*/HH,
            bsum0, c0, h0b, (__bf16*)nullptr);
        // layer 1: A = [h0_new | h1], W = [W_ih1 | W_hh1]; h1 -> feed + outs[t]
        lstm_step_kernel<<<32, 256, 0, stream>>>(
            (const float*)nullptr, /*xK=*/0, /*aB0=*/h0b, /*aB1=*/h1b, /*KTOT=*/2 * HH,
            Wih1b, /*ws0=*/HH, /*wsplit=*/HH, Whh1b, /*ws1=*/HH,
            bsum1, c1, h1b, outsb + (size_t)t * BB * HH);
    }

    // ---- 4) projection + bias + residual, then LayerNorm (in place on d_out)
    proj_kernel<<<dim3(HH / 128, TT), 256, 0, stream>>>(outsb, Wlinb, b_lin, x, y);
    ln_kernel<<<(TT * BB) / 8, 256, 0, stream>>>(y, ln_g, ln_b, TT * BB);
}
